// RGTransformer_103079215838
// MI455X (gfx1250) — matrix operand order repaired and checked by reading, not compiled, not run on hardware
//
#include <hip/hip_runtime.h>
#include <hip/hip_bf16.h>

typedef __bf16 bf16_t;
typedef __attribute__((ext_vector_type(16))) __bf16 v16bf;
typedef __attribute__((ext_vector_type(8)))  __bf16 v8bf;
typedef __attribute__((ext_vector_type(8)))  float  v8f;

#define WMMA_BF16(a, b, c) \
  __builtin_amdgcn_wmma_f32_16x16x32_bf16(false, (a), false, (b), (short)0, (c), false, false)

__device__ __forceinline__ bf16_t f2bf(float x) { return (bf16_t)x; }

// ---- Fragment loaders (16-bit WMMA operands, contiguous k order for both A and B) -------------
// A 16x32: lane holds row m = lane%16; lanes 0-15: K 0..15; lanes 16-31: K 16..31 (contiguous)
__device__ __forceinline__ v16bf load_a_frag(const bf16_t* __restrict__ row, int kbase, int lane) {
  const int sel = (lane & 16) ? 16 : 0;
  union { v16bf v; v8bf h[2]; } u;
  u.h[0] = *(const v8bf*)(row + kbase + sel);
  u.h[1] = *(const v8bf*)(row + kbase + sel + 8);
  return u.v;
}
// B 32x16: lane holds column n = lane%16; lanes 0-15: K 0..15; lanes 16-31: K 16..31 (contiguous)
__device__ __forceinline__ v16bf load_b_frag(const bf16_t* __restrict__ colrow, int kbase, int lane) {
  const int sel = (lane & 16) ? 16 : 0;
  return *(const v16bf*)(colrow + kbase + sel);
}

// ---- LayerNorm + bf16 cast: one wave (32 lanes) per row of 256 ------------
__global__ void ln_cast_kernel(const float* __restrict__ X,
                               const float* __restrict__ g, const float* __restrict__ b,
                               bf16_t* __restrict__ lnout, bf16_t* __restrict__ xbout) {
  const int row  = blockIdx.x;
  const int lane = threadIdx.x;
  const float* xr = X + (size_t)row * 256;
  float v[8]; float s = 0.f;
#pragma unroll
  for (int i = 0; i < 8; ++i) { v[i] = xr[lane + 32 * i]; s += v[i]; }
#pragma unroll
  for (int o = 16; o > 0; o >>= 1) s += __shfl_xor(s, o, 32);
  const float mean = s * (1.f / 256.f);
  float vs = 0.f;
#pragma unroll
  for (int i = 0; i < 8; ++i) { float d = v[i] - mean; vs += d * d; }
#pragma unroll
  for (int o = 16; o > 0; o >>= 1) vs += __shfl_xor(vs, o, 32);
  const float rstd = rsqrtf(vs * (1.f / 256.f) + 1e-5f);
#pragma unroll
  for (int i = 0; i < 8; ++i) {
    const int c = lane + 32 * i;
    const float y = (v[i] - mean) * rstd * g[c] + b[c];
    lnout[(size_t)row * 256 + c] = f2bf(y);
    xbout[(size_t)row * 256 + c] = f2bf(v[i]);
  }
}

// ---- Weight transpose + cast: Wt[n,k] = bf16(W[k,n]), 256x256 -------------
__global__ void wt_kernel(const float* __restrict__ W, bf16_t* __restrict__ Wt) {
  const int n = blockIdx.x, k = threadIdx.x;
  Wt[(size_t)n * 256 + k] = f2bf(W[(size_t)k * 256 + n]);
}

// ---- bf16 WMMA GEMM: C[M,N] = A[M,256] @ Wt^T + bias ----------------------
// MODE 0: bf16 out; 1: f32 out; 2: f32 sigmoid out.
// Wave computes a 32x64 tile: 2 A-frags x 4 B-frags -> 8 WMMAs per K-step.
template <int MODE>
__global__ __launch_bounds__(128) void gemm_bf16(const bf16_t* __restrict__ A,
                                                 const bf16_t* __restrict__ Wt,
                                                 const float* __restrict__ bias,
                                                 void* __restrict__ Out, int N, int K) {
  const int lane  = threadIdx.x & 31;
  const int wave  = threadIdx.x >> 5;
  const int mbase = blockIdx.x * 32;
  const int nbase = wave * 64;
  v8f acc[2][4] = {};
  const bf16_t* arow0 = A + (size_t)(mbase + (lane & 15)) * K;
  const bf16_t* arow1 = arow0 + (size_t)16 * K;
#pragma unroll
  for (int k = 0; k < 256; k += 32) {
    const v16bf a0 = load_a_frag(arow0, k, lane);
    const v16bf a1 = load_a_frag(arow1, k, lane);
#pragma unroll
    for (int j = 0; j < 4; ++j) {
      const bf16_t* wrow = Wt + (size_t)(nbase + j * 16 + (lane & 15)) * K;
      const v16bf bfr = load_b_frag(wrow, k, lane);
      acc[0][j] = WMMA_BF16(a0, bfr, acc[0][j]);
      acc[1][j] = WMMA_BF16(a1, bfr, acc[1][j]);
    }
  }
#pragma unroll
  for (int i = 0; i < 2; ++i) {
    const int mrow0 = mbase + i * 16 + ((lane & 16) ? 8 : 0);
#pragma unroll
    for (int j = 0; j < 4; ++j) {
      const int n  = nbase + j * 16 + (lane & 15);
      const float bv = bias[n];
#pragma unroll
      for (int r = 0; r < 8; ++r) {
        float x = acc[i][j][r] + bv;
        const size_t idx = (size_t)(mrow0 + r) * N + n;
        if (MODE == 0) {
          ((bf16_t*)Out)[idx] = f2bf(x);
        } else {
          if (MODE == 2) x = 1.0f / (1.0f + __expf(-x));
          ((float*)Out)[idx] = x;
        }
      }
    }
  }
}

// ---- V -> V^T [B,H,DH,S] bf16 ---------------------------------------------
__global__ void vt_kernel(const bf16_t* __restrict__ V, bf16_t* __restrict__ vT) {
  const size_t i = (size_t)blockIdx.x * blockDim.x + threadIdx.x;  // over B*S*E
  const int e = (int)(i & 255);
  const int s = (int)((i >> 8) & 1023);
  const int b = (int)(i >> 18);
  const int h = e >> 5, d = e & 31;
  vT[(((size_t)(b * 8 + h) * 32 + d) << 10) + s] = V[i];
}

// ---- Flash attention: one wave per 16-row q tile per (b,h) ----------------
// 64-key chunks: 4 score WMMAs + 4 AV WMMAs per chunk; deferred row-sum
// reduction (per-lane partials, one 16-lane reduce at the end).
__global__ __launch_bounds__(128) void attn_kernel(const bf16_t* __restrict__ Q,
                                                   const bf16_t* __restrict__ Kb,
                                                   const bf16_t* __restrict__ vT,
                                                   const float* __restrict__ rel_bias,
                                                   bf16_t* __restrict__ O) {
  const int lane = threadIdx.x & 31;
  const int wave = threadIdx.x >> 5;
  const int tile = blockIdx.x * 4 + wave;  // B*H*(S/16) = 8192 tiles
  const int qt = tile & 63;
  const int h  = (tile >> 6) & 7;
  const int b  = tile >> 9;
  const int qbase = qt * 16;
  const int ncol = lane & 15;
  const int rrow0 = (lane & 16) ? 8 : 0;

  __shared__ __align__(64) bf16_t stage[4][16 * 64];
  bf16_t* sm = stage[wave];

  // q A-fragment (reused for every key chunk)
  const bf16_t* qrow = Q + ((size_t)(b * 1024 + qbase + ncol) * 256) + h * 32;
  const v16bf aq = load_a_frag(qrow, 0, lane);

  const float scale = 0.1767766952966369f;  // 1/sqrt(DH=32)
  float mrow[8], lsp[8];  // running max + per-lane partial sums
  v8f oacc0 = {}, oacc1 = {};
#pragma unroll
  for (int r = 0; r < 8; ++r) { mrow[r] = -1e30f; lsp[r] = 0.f; }
  const float* rb_base = rel_bias + (size_t)(qbase + rrow0) * 1024;
  const bf16_t* vbase = vT + (((size_t)(b * 8 + h) * 32) << 10);

  for (int kb = 0; kb < 1024; kb += 64) {
    // ---- scores: four 16x16 tiles (columns kb+16t+ncol) ----
    v8f s[4];
    {
      const int sel = (lane & 16) ? 16 : 0;  // dh half per half-wave
#pragma unroll
      for (int t = 0; t < 4; ++t) {
        const bf16_t* krow = Kb + ((size_t)(b * 1024 + kb + t * 16 + ncol) * 256) + h * 32;
        const v16bf bk = *(const v16bf*)(krow + sel);
        v8f z = {};
        s[t] = WMMA_BF16(aq, bk, z);
      }
      if (kb + 64 < 1024) {  // prefetch next chunk (global_prefetch_b8)
        __builtin_prefetch(Kb + ((size_t)(b * 1024 + kb + 64 + ncol) * 256) + h * 32, 0, 0);
        __builtin_prefetch(vbase + ((size_t)ncol << 10) + kb + 64, 0, 0);
        __builtin_prefetch(vbase + ((size_t)(16 + ncol) << 10) + kb + 64, 0, 0);
      }
    }
    // scale + relational bias; local max over the 4 tiles
    float pmax[8];
#pragma unroll
    for (int r = 0; r < 8; ++r) {
      const float* rb = rb_base + (size_t)r * 1024 + kb + ncol;
      const float x0 = s[0][r] * scale + rb[0];
      const float x1 = s[1][r] * scale + rb[16];
      const float x2 = s[2][r] * scale + rb[32];
      const float x3 = s[3][r] * scale + rb[48];
      s[0][r] = x0; s[1][r] = x1; s[2][r] = x2; s[3][r] = x3;
      pmax[r] = fmaxf(fmaxf(x0, x1), fmaxf(x2, x3));
    }
#pragma unroll
    for (int o = 1; o < 16; o <<= 1)
#pragma unroll
      for (int r = 0; r < 8; ++r) pmax[r] = fmaxf(pmax[r], __shfl_xor(pmax[r], o, 32));
    // online-softmax update; exp -> LDS (C layout -> A layout re-staging)
#pragma unroll
    for (int r = 0; r < 8; ++r) {
      const float mnew = fmaxf(mrow[r], pmax[r]);
      const float corr = __expf(mrow[r] - mnew);
      mrow[r] = mnew;
      const float e0 = __expf(s[0][r] - mnew);
      const float e1 = __expf(s[1][r] - mnew);
      const float e2 = __expf(s[2][r] - mnew);
      const float e3 = __expf(s[3][r] - mnew);
      bf16_t* smr = sm + (rrow0 + r) * 64 + ncol;
      smr[0]  = f2bf(e0);
      smr[16] = f2bf(e1);
      smr[32] = f2bf(e2);
      smr[48] = f2bf(e3);
      lsp[r] = lsp[r] * corr + ((e0 + e1) + (e2 + e3));
      oacc0[r] *= corr;
      oacc1[r] *= corr;
    }
    // attn @ V : two K=32 halves x two DH tiles (same-wave DS ops in-order)
#pragma unroll
    for (int half = 0; half < 2; ++half) {
      v16bf ap;
      {
        const int selr = (lane & 16) ? 16 : 0;
        const bf16_t* base = sm + ncol * 64 + half * 32;
        union { v16bf v; v8bf hh[2]; } u;
        u.hh[0] = *(const v8bf*)(base + selr);
        u.hh[1] = *(const v8bf*)(base + selr + 8);
        ap = u.v;
      }
      const int sel = (lane & 16) ? 16 : 0;  // k half per half-wave
      const v16bf bv0 = *(const v16bf*)(vbase + ((size_t)ncol << 10) + kb + half * 32 + sel);
      const v16bf bv1 = *(const v16bf*)(vbase + ((size_t)(16 + ncol) << 10) + kb + half * 32 + sel);
      oacc0 = WMMA_BF16(ap, bv0, oacc0);
      oacc1 = WMMA_BF16(ap, bv1, oacc1);
    }
  }
  // deferred row-sum reduction (once, not per chunk)
#pragma unroll
  for (int o = 1; o < 16; o <<= 1)
#pragma unroll
    for (int r = 0; r < 8; ++r) lsp[r] += __shfl_xor(lsp[r], o, 32);
  // finalize: normalize and store o (bf16) at [b, q, h*32 + d]
#pragma unroll
  for (int r = 0; r < 8; ++r) {
    const float inv = 1.0f / lsp[r];
    bf16_t* orow = O + ((size_t)(b * 1024 + qbase + rrow0 + r) * 256) + h * 32;
    orow[ncol]      = f2bf(oacc0[r] * inv);
    orow[16 + ncol] = f2bf(oacc1[r] * inv);
  }
}

// ---- gate1 = sigmoid_gate * out1 + Matrix (f32 + bf16 copies) -------------
__global__ void gate_kernel(const float* __restrict__ g0, const float* __restrict__ o1,
                            const float* __restrict__ X, float* __restrict__ g1f,
                            bf16_t* __restrict__ g1b) {
  const size_t i = (size_t)blockIdx.x * blockDim.x + threadIdx.x;
  const float v = g0[i] * o1[i] + X[i];
  g1f[i] = v;
  g1b[i] = f2bf(v);
}

// ---- final = gate1 * layernorm(mlp) + gate1 : one wave per row ------------
__global__ void final_kernel(const float* __restrict__ mlp, const float* __restrict__ g1,
                             const float* __restrict__ g, const float* __restrict__ b,
                             float* __restrict__ out) {
  const int row  = blockIdx.x;
  const int lane = threadIdx.x;
  const float* mr = mlp + (size_t)row * 256;
  float v[8]; float s = 0.f;
#pragma unroll
  for (int i = 0; i < 8; ++i) { v[i] = mr[lane + 32 * i]; s += v[i]; }
#pragma unroll
  for (int o = 16; o > 0; o >>= 1) s += __shfl_xor(s, o, 32);
  const float mean = s * (1.f / 256.f);
  float vs = 0.f;
#pragma unroll
  for (int i = 0; i < 8; ++i) { float d = v[i] - mean; vs += d * d; }
#pragma unroll
  for (int o = 16; o > 0; o >>= 1) vs += __shfl_xor(vs, o, 32);
  const float rstd = rsqrtf(vs * (1.f / 256.f) + 1e-5f);
#pragma unroll
  for (int i = 0; i < 8; ++i) {
    const int c = lane + 32 * i;
    const float ln = (v[i] - mean) * rstd * g[c] + b[c];
    const float gv = g1[(size_t)row * 256 + c];
    out[(size_t)row * 256 + c] = gv * ln + gv;
  }
}

extern "C" void kernel_launch(void* const* d_in, const int* in_sizes, int n_in,
                              void* d_out, int out_size, void* d_ws, size_t ws_size,
                              hipStream_t stream) {
  (void)in_sizes; (void)n_in; (void)out_size; (void)ws_size;
  const float* Matrix   = (const float*)d_in[0];
  const float* ln_g     = (const float*)d_in[1];
  const float* ln_b     = (const float*)d_in[2];
  const float* Wq       = (const float*)d_in[3];
  const float* bq       = (const float*)d_in[4];
  const float* Wk       = (const float*)d_in[5];
  const float* bk       = (const float*)d_in[6];
  const float* Wv       = (const float*)d_in[7];
  const float* bv       = (const float*)d_in[8];
  const float* Wo       = (const float*)d_in[9];
  const float* bo       = (const float*)d_in[10];
  const float* rel_bias = (const float*)d_in[11];
  const float* gW       = (const float*)d_in[12];
  const float* gb       = (const float*)d_in[13];
  const float* mW       = (const float*)d_in[14];
  const float* mb       = (const float*)d_in[15];
  float* out = (float*)d_out;

  char* ws = (char*)d_ws;
  const size_t MB = 1ull << 20;
  bf16_t* ln    = (bf16_t*)(ws + 0 * MB);    // 8 MB
  bf16_t* xb    = (bf16_t*)(ws + 8 * MB);    // 8 MB
  bf16_t* Qb    = (bf16_t*)(ws + 16 * MB);   // 8 MB
  bf16_t* Kbuf  = (bf16_t*)(ws + 24 * MB);   // 8 MB
  bf16_t* Vb    = (bf16_t*)(ws + 32 * MB);   // 8 MB
  bf16_t* vT    = (bf16_t*)(ws + 40 * MB);   // 8 MB
  float*  gate0 = (float*)(ws + 48 * MB);    // 16 MB
  bf16_t* Ob    = (bf16_t*)(ws + 64 * MB);   // 8 MB
  bf16_t* Wt0   = (bf16_t*)(ws + 72 * MB);   // 6 x 128 KB
  bf16_t* Wt1   = Wt0 + 1 * 65536;
  bf16_t* Wt2   = Wt0 + 2 * 65536;
  bf16_t* Wt3   = Wt0 + 3 * 65536;
  bf16_t* Wt4   = Wt0 + 4 * 65536;
  bf16_t* Wt5   = Wt0 + 5 * 65536;
  // phase-2 aliases (earlier buffers dead by then)
  float*  out1  = (float*)(ws + 0 * MB);     // over ln+xb (16 MB)
  float*  mlpf  = (float*)(ws + 16 * MB);    // over Qb+Kbuf (16 MB)
  bf16_t* g1b   = (bf16_t*)(ws + 32 * MB);   // over Vb (8 MB)
  float*  g1f   = gate0;                     // in-place over gate0

  const int ROWS = 16 * 1024;  // B*S

  ln_cast_kernel<<<ROWS, 32, 0, stream>>>(Matrix, ln_g, ln_b, ln, xb);

  wt_kernel<<<256, 256, 0, stream>>>(Wq, Wt0);
  wt_kernel<<<256, 256, 0, stream>>>(Wk, Wt1);
  wt_kernel<<<256, 256, 0, stream>>>(Wv, Wt2);
  wt_kernel<<<256, 256, 0, stream>>>(Wo, Wt3);
  wt_kernel<<<256, 256, 0, stream>>>(gW, Wt4);
  wt_kernel<<<256, 256, 0, stream>>>(mW, Wt5);

  gemm_bf16<0><<<ROWS / 32, 128, 0, stream>>>(ln, Wt0, bq, (void*)Qb, 256, 256);
  gemm_bf16<0><<<ROWS / 32, 128, 0, stream>>>(ln, Wt1, bk, (void*)Kbuf, 256, 256);
  gemm_bf16<0><<<ROWS / 32, 128, 0, stream>>>(ln, Wt2, bv, (void*)Vb, 256, 256);
  gemm_bf16<2><<<ROWS / 32, 128, 0, stream>>>(xb, Wt4, gb, (void*)gate0, 256, 256);

  vt_kernel<<<(16 * 1024 * 256) / 256, 256, 0, stream>>>(Vb, vT);

  attn_kernel<<<2048, 128, 0, stream>>>(Qb, Kbuf, vT, rel_bias, Ob);

  gemm_bf16<1><<<ROWS / 32, 128, 0, stream>>>(Ob, Wt3, bo, (void*)out1, 256, 256);

  gate_kernel<<<(16 * 1024 * 256) / 256, 256, 0, stream>>>(gate0, out1, Matrix, g1f, g1b);

  gemm_bf16<2><<<ROWS / 32, 128, 0, stream>>>(g1b, Wt5, mb, (void*)mlpf, 256, 256);

  final_kernel<<<ROWS, 32, 0, stream>>>(mlpf, g1f, ln_g, ln_b, out);
}
